// CustomBigBirdForSequenceClassification_24455543783885
// MI455X (gfx1250) — matrix operand order repaired and checked by reading, not compiled, not run on hardware
//
#include <hip/hip_runtime.h>
#include <hip/hip_bf16.h>
#include <math.h>

// ---------------------------------------------------------------------------
// Types for CDNA5 WMMA (wave32): v_wmma_f32_16x16x32_bf16
// ---------------------------------------------------------------------------
typedef __bf16 bf16;
typedef __attribute__((ext_vector_type(16))) __bf16 v16bf;
typedef __attribute__((ext_vector_type(8)))  float  v8f;
typedef __attribute__((ext_vector_type(4)))  __bf16 bf16x4;

// Async global->LDS copy path (gfx1250 GLOBAL_LOAD_ASYNC_TO_LDS_B128, ASYNCcnt)
#if __has_builtin(__builtin_amdgcn_global_load_async_to_lds_b128) && \
    __has_builtin(__builtin_amdgcn_s_wait_asynccnt)
#define USE_ASYNC_LDS 1
typedef __attribute__((__vector_size__(4 * sizeof(int)))) int i32x4;
typedef __attribute__((address_space(1))) i32x4 g_i32x4;
typedef __attribute__((address_space(3))) i32x4 l_i32x4;
// copy 16 bytes (8 bf16) from global to LDS, per-lane addresses
__device__ inline void async_cp16(bf16* dst, const bf16* src) {
  __builtin_amdgcn_global_load_async_to_lds_b128(
      (g_i32x4*)(unsigned long long)src,
      (l_i32x4*)(unsigned)(unsigned long long)dst, 0, 0);
}
__device__ inline void async_wait() { __builtin_amdgcn_s_wait_asynccnt(0); }
#else
#define USE_ASYNC_LDS 0
__device__ inline void async_cp16(bf16* dst, const bf16* src) {
  *(bf16x4*)dst       = *(const bf16x4*)src;
  *(bf16x4*)(dst + 4) = *(const bf16x4*)(src + 4);
}
__device__ inline void async_wait() {}
#endif

__device__ inline v8f wmma_bf16(v16bf a, v16bf b, v8f c) {
  // D = A(16x32 bf16) x B(32x16 bf16) + C(16x16 f32)
  return __builtin_amdgcn_wmma_f32_16x16x32_bf16(
      /*neg_a=*/false, a, /*neg_b=*/false, b,
      /*c_mod=*/(short)0, c, /*reuse_a=*/false, /*reuse_b=*/false);
}

// A fragment: 16x32 bf16 tile at src (row stride ld).
// lanes 0-15 (hi=0): K in {0..7, 16..23}; lanes 16-31: K in {8..15, 24..31}
// Two contiguous 8-element runs per lane -> 2x ds_load_b128.
__device__ inline v16bf load_frag_a(const bf16* src, int ld) {
  int lane = threadIdx.x & 31;
  int row = lane & 15, hi = lane >> 4;
  v16bf a;
#pragma unroll
  for (int e = 0; e < 16; ++e) {
    int k = e + hi * 8 + ((e >= 8) ? 8 : 0);
    a[e] = src[row * ld + k];
  }
  return a;
}

// B fragment from a K-contiguous source: logical B[k][n] = src[n*ld + k].
// Per-lane K in {8hi..8hi+7, 16+8hi..16+8hi+7} (matches A) -> 2x ds_load_b128.
__device__ inline v16bf load_frag_bT(const bf16* src, int ld) {
  int lane = threadIdx.x & 31;
  int col = lane & 15, hi = lane >> 4;
  v16bf b;
#pragma unroll
  for (int e = 0; e < 16; ++e) b[e] = src[col * ld + e + hi * 8 + ((e >= 8) ? 8 : 0)];
  return b;
}

// ---------------------------------------------------------------------------
// Weight prep (once per launch): fp32 [K][N] -> bf16 transposed [N][K]
// ---------------------------------------------------------------------------
__global__ void cvt_transpose_f32_to_bf16(const float* __restrict__ src,
                                          bf16* __restrict__ dst, int K, int N) {
  int i = blockIdx.x * blockDim.x + threadIdx.x;   // over N*K
  if (i < N * K) {
    int n = i / K, k = i - n * K;
    dst[i] = (bf16)src[(size_t)k * N + n];
  }
}

// ---------------------------------------------------------------------------
// Embedding gather + LayerNorm  (one row of S per block)
// ---------------------------------------------------------------------------
__global__ __launch_bounds__(256) void embed_ln_kernel(
    const int* __restrict__ ids, const float* __restrict__ wemb,
    const float* __restrict__ pemb, const float* __restrict__ g,
    const float* __restrict__ b, float* __restrict__ hf, bf16* __restrict__ hb) {
  __shared__ float sx[768];
  __shared__ float red[256];
  int s = blockIdx.x, t = threadIdx.x;
  int id = ids[s];
  float lsum = 0.f;
  for (int c = t; c < 768; c += 256) {
    float x = wemb[(size_t)id * 768 + c] + pemb[(size_t)s * 768 + c];
    sx[c] = x; lsum += x;
  }
  red[t] = lsum; __syncthreads();
  for (int o = 128; o > 0; o >>= 1) { if (t < o) red[t] += red[t + o]; __syncthreads(); }
  float mu = red[0] * (1.0f / 768.0f); __syncthreads();
  float lvar = 0.f;
  for (int c = t; c < 768; c += 256) { float d = sx[c] - mu; lvar += d * d; }
  red[t] = lvar; __syncthreads();
  for (int o = 128; o > 0; o >>= 1) { if (t < o) red[t] += red[t + o]; __syncthreads(); }
  float inv = rsqrtf(red[0] * (1.0f / 768.0f) + 1e-12f);
  for (int c = t; c < 768; c += 256) {
    float y = (sx[c] - mu) * inv * g[c] + b[c];
    hf[(size_t)s * 768 + c] = y;
    hb[(size_t)s * 768 + c] = (bf16)y;
  }
}

// ---------------------------------------------------------------------------
// Residual add + LayerNorm (in-place safe: row fully read into LDS first)
// ---------------------------------------------------------------------------
__global__ __launch_bounds__(256) void add_ln_kernel(
    const float* __restrict__ hin, const float* __restrict__ a,
    const float* __restrict__ g, const float* __restrict__ b,
    float* __restrict__ hf, bf16* __restrict__ hb) {
  __shared__ float sx[768];
  __shared__ float red[256];
  int s = blockIdx.x, t = threadIdx.x;
  float lsum = 0.f;
  for (int c = t; c < 768; c += 256) {
    float x = hin[(size_t)s * 768 + c] + a[(size_t)s * 768 + c];
    sx[c] = x; lsum += x;
  }
  red[t] = lsum; __syncthreads();
  for (int o = 128; o > 0; o >>= 1) { if (t < o) red[t] += red[t + o]; __syncthreads(); }
  float mu = red[0] * (1.0f / 768.0f); __syncthreads();
  float lvar = 0.f;
  for (int c = t; c < 768; c += 256) { float d = sx[c] - mu; lvar += d * d; }
  red[t] = lvar; __syncthreads();
  for (int o = 128; o > 0; o >>= 1) { if (t < o) red[t] += red[t + o]; __syncthreads(); }
  float inv = rsqrtf(red[0] * (1.0f / 768.0f) + 1e-12f);
  for (int c = t; c < 768; c += 256) {
    float y = (sx[c] - mu) * inv * g[c] + b[c];
    hf[(size_t)s * 768 + c] = y;
    hb[(size_t)s * 768 + c] = (bf16)y;
  }
}

// ---------------------------------------------------------------------------
// WMMA GEMM: out = act(A[M,K]bf16 @ B + bias), B given TRANSPOSED: Wt[N][K].
// Workgroup tile 64x64, 8 waves, 2 subtiles per wave (B fragment reused).
// Staging via async global->LDS b128 copies; fragments load as ds_load_b128.
// outBT (optional) writes the result transposed: [N][M] bf16.
// ---------------------------------------------------------------------------
__global__ __launch_bounds__(256) void gemm_bf16_kernel(
    const bf16* __restrict__ A, const bf16* __restrict__ Wt,
    const float* __restrict__ bias, float* __restrict__ outF,
    bf16* __restrict__ outB, bf16* __restrict__ outBT,
    int M, int N, int K, int act_gelu) {
  __shared__ __align__(16) bf16 sA [64 * 32];   // row-major [m][k]
  __shared__ __align__(16) bf16 sBt[64 * 32];   // col-major [n][k]
  const int t = threadIdx.x;
  const int wave = t >> 5;
  const int m0 = blockIdx.x * 64;
  const int n0 = blockIdx.y * 64;
  const int mr0 = wave >> 2;        // 0..1 ; second subtile uses mr0+2
  const int nc  = wave & 3;         // shared by both subtiles
  const int i8  = t * 8;            // this thread's 8-element (16B) chunk
  const int sr  = i8 >> 5, scol = i8 & 31;
  v8f acc0 = {}, acc1 = {};
  for (int k0 = 0; k0 < K; k0 += 32) {
    __syncthreads();
    async_cp16(&sA [i8], &A [(size_t)(m0 + sr) * K + k0 + scol]);
    async_cp16(&sBt[i8], &Wt[(size_t)(n0 + sr) * K + k0 + scol]);
    if (k0 + 32 < K) {   // global_prefetch_b8 of next chunks
      __builtin_prefetch(&A [(size_t)(m0 + sr) * K + k0 + 32], 0, 3);
      __builtin_prefetch(&Wt[(size_t)(n0 + sr) * K + k0 + 32], 0, 3);
    }
    async_wait();
    __syncthreads();
    v16bf b  = load_frag_bT(&sBt[(nc * 16) * 32], 32);
    v16bf a0 = load_frag_a (&sA [(mr0 * 16) * 32], 32);
    v16bf a1 = load_frag_a (&sA [((mr0 + 2) * 16) * 32], 32);
    acc0 = wmma_bf16(a0, b, acc0);
    acc1 = wmma_bf16(a1, b, acc1);
  }
  int lane = t & 31, col = lane & 15, hi = lane >> 4;
#pragma unroll
  for (int si = 0; si < 2; ++si) {
    int mr = mr0 + si * 2;
    v8f acc = si ? acc1 : acc0;
#pragma unroll
    for (int r = 0; r < 8; ++r) {
      int gm = m0 + mr * 16 + r + hi * 8;
      int gn = n0 + nc * 16 + col;
      float v = acc[r] + bias[gn];
      if (act_gelu) {
        float x = v;
        v = 0.5f * x * (1.0f + tanhf(0.7978845608028654f * (x + 0.044715f * x * x * x)));
      }
      if (outF)  outF [(size_t)gm * N + gn] = v;
      if (outB)  outB [(size_t)gm * N + gn] = (bf16)v;
      if (outBT) outBT[(size_t)gn * M + gm] = (bf16)v;
    }
  }
}

// ---------------------------------------------------------------------------
// BigBird block list for middle q-blocks: {0, 63, n-1, n, n+1, r0, r1, r2}
// ---------------------------------------------------------------------------
__device__ inline void make_blocklist(const int* __restrict__ rand_attn,
                                      int layer, int h, int n, int* list) {
  list[0] = 0; list[1] = 63; list[2] = n - 1; list[3] = n; list[4] = n + 1;
  const int* r = rand_attn + (((size_t)layer * 12 + h) * 64 + n) * 3;
  list[5] = r[0]; list[6] = r[1]; list[7] = r[2];
}

// ---------------------------------------------------------------------------
// Scores: S = (Q @ K^T) * 1/sqrt(64), stored bf16.
// One block per (head, q-block, chunk of 8 k-blocks).
// ---------------------------------------------------------------------------
__global__ __launch_bounds__(256) void attn_scores_kernel(
    const bf16* __restrict__ qb, const bf16* __restrict__ kb,
    const int* __restrict__ rand_attn, int layer,
    bf16* __restrict__ sc_mid, bf16* __restrict__ sc_edge) {
  __shared__ __align__(16) bf16 sQ[64 * 64];
  __shared__ __align__(16) bf16 sK[64 * 64];
  int h = blockIdx.x, n = blockIdx.y, z = blockIdx.z;
  bool edge = (n == 0) || (n == 63);
  if (!edge && z != 0) return;
  int list[8];
  if (!edge) make_blocklist(rand_attn, layer, h, n, list);
  const int W = edge ? 4096 : 512;
  bf16* dst = edge ? sc_edge + (((size_t)h * 2 + (n ? 1 : 0)) * 64) * 4096
                   : sc_mid  + (((size_t)h * 62 + (n - 1)) * 64) * 512;
  int t = threadIdx.x, wave = t >> 5;
  const int i16 = t * 16;                       // 16-elem chunk (two 16B copies)
  const int qr0 = i16 >> 6, qc0 = i16 & 63;
  async_cp16(&sQ[i16],     &qb[((size_t)(n * 64 + qr0)) * 768 + h * 64 + qc0]);
  async_cp16(&sQ[i16 + 8], &qb[((size_t)(n * 64 + qr0)) * 768 + h * 64 + qc0 + 8]);
  for (int tt = 0; tt < 8; ++tt) {
    int jblk = edge ? z * 8 + tt : list[tt];
    int dcol = edge ? (z * 8 + tt) * 64 : tt * 64;
    __syncthreads();
    async_cp16(&sK[i16],     &kb[((size_t)(jblk * 64 + qr0)) * 768 + h * 64 + qc0]);
    async_cp16(&sK[i16 + 8], &kb[((size_t)(jblk * 64 + qr0)) * 768 + h * 64 + qc0 + 8]);
    async_wait();
    __syncthreads();
#pragma unroll
    for (int si = 0; si < 2; ++si) {              // 2 subtiles per wave
      int st = wave + si * 8;
      int mr = st >> 2, nc = st & 3;
      v8f acc = {};
#pragma unroll
      for (int ks = 0; ks < 2; ++ks) {
        v16bf a = load_frag_a (&sQ[(mr * 16) * 64 + ks * 32], 64);
        v16bf b = load_frag_bT(&sK[(nc * 16) * 64 + ks * 32], 64);
        acc = wmma_bf16(a, b, acc);
      }
      int lane = t & 31, col = lane & 15, hi = lane >> 4;
#pragma unroll
      for (int r = 0; r < 8; ++r) {
        int qr = mr * 16 + r + hi * 8;
        dst[(size_t)qr * W + dcol + nc * 16 + col] = (bf16)(acc[r] * 0.125f);
      }
    }
  }
}

// ---------------------------------------------------------------------------
// Row softmax over bf16 scores, in place (read-before-write per element).
// One wave per row; vectorized bf16x4 lane accesses; wave32 shuffles.
// ---------------------------------------------------------------------------
__global__ __launch_bounds__(256) void softmax_kernel(bf16* sc_mid, bf16* sc_edge) {
  int h = blockIdx.x, n = blockIdx.y, rg = blockIdx.z;
  int wave = threadIdx.x >> 5, lane = threadIdx.x & 31;
  int r = rg * 8 + wave;
  bool edge = (n == 0) || (n == 63);
  int W = edge ? 4096 : 512;
  bf16* row = edge ? sc_edge + (((size_t)h * 2 + (n ? 1 : 0)) * 64 + r) * 4096
                   : sc_mid  + (((size_t)h * 62 + (n - 1)) * 64 + r) * 512;
  float m = -1e30f;
  for (int i = lane * 4; i < W; i += 128) {
    bf16x4 v4 = *(const bf16x4*)&row[i];
#pragma unroll
    for (int q = 0; q < 4; ++q) m = fmaxf(m, (float)v4[q]);
  }
  for (int o = 16; o > 0; o >>= 1) m = fmaxf(m, __shfl_xor(m, o, 32));
  float sum = 0.f;
  for (int i = lane * 4; i < W; i += 128) {
    bf16x4 v4 = *(const bf16x4*)&row[i];
#pragma unroll
    for (int q = 0; q < 4; ++q) sum += __expf((float)v4[q] - m);
  }
  for (int o = 16; o > 0; o >>= 1) sum += __shfl_xor(sum, o, 32);
  float inv = 1.f / sum;
  for (int i = lane * 4; i < W; i += 128) {
    bf16x4 v4 = *(const bf16x4*)&row[i];
#pragma unroll
    for (int q = 0; q < 4; ++q) v4[q] = (bf16)(__expf((float)v4[q] - m) * inv);
    *(bf16x4*)&row[i] = v4;
  }
}

// ---------------------------------------------------------------------------
// O = P @ V over the block list, per (head, q-block). Output bf16 [S, H*64].
// P is compact bf16; V comes pre-transposed ([h*64+d][s]) from the V GEMM,
// so both LDS stages are contiguous async b128 copies.
// ---------------------------------------------------------------------------
__global__ __launch_bounds__(256) void attn_pv_kernel(
    const bf16* __restrict__ sc_mid, const bf16* __restrict__ sc_edge,
    const bf16* __restrict__ vt, const int* __restrict__ rand_attn, int layer,
    bf16* __restrict__ ob) {
  __shared__ __align__(16) bf16 sP [64 * 64];   // [q][key]
  __shared__ __align__(16) bf16 sVt[64 * 64];   // [d][key]
  int h = blockIdx.x, n = blockIdx.y;
  bool edge = (n == 0) || (n == 63);
  int Wn = edge ? 64 : 8;
  int W  = edge ? 4096 : 512;
  const bf16* pbase = edge ? sc_edge + (((size_t)h * 2 + (n ? 1 : 0)) * 64) * 4096
                           : sc_mid  + (((size_t)h * 62 + (n - 1)) * 64) * 512;
  int list[8];
  if (!edge) make_blocklist(rand_attn, layer, h, n, list);
  int t = threadIdx.x, wave = t >> 5;
  const int mr0 = wave >> 2, nc = wave & 3;
  const int i16 = t * 16;
  const int r0 = i16 >> 6, c0 = i16 & 63;
  v8f acc0 = {}, acc1 = {};
  for (int tt = 0; tt < Wn; ++tt) {
    int jblk = edge ? tt : list[tt];
    __syncthreads();
    async_cp16(&sP[i16],      &pbase[(size_t)r0 * W + tt * 64 + c0]);
    async_cp16(&sP[i16 + 8],  &pbase[(size_t)r0 * W + tt * 64 + c0 + 8]);
    async_cp16(&sVt[i16],     &vt[((size_t)(h * 64 + r0)) * 4096 + jblk * 64 + c0]);
    async_cp16(&sVt[i16 + 8], &vt[((size_t)(h * 64 + r0)) * 4096 + jblk * 64 + c0 + 8]);
    async_wait();
    __syncthreads();
#pragma unroll
    for (int ks = 0; ks < 2; ++ks) {
      v16bf b  = load_frag_bT(&sVt[(nc * 16) * 64 + ks * 32], 64);
      v16bf a0 = load_frag_a (&sP [(mr0 * 16) * 64 + ks * 32], 64);
      v16bf a1 = load_frag_a (&sP [((mr0 + 2) * 16) * 64 + ks * 32], 64);
      acc0 = wmma_bf16(a0, b, acc0);
      acc1 = wmma_bf16(a1, b, acc1);
    }
  }
  int lane = t & 31, col = lane & 15, hi = lane >> 4;
#pragma unroll
  for (int si = 0; si < 2; ++si) {
    int mr = mr0 + si * 2;
    v8f acc = si ? acc1 : acc0;
#pragma unroll
    for (int r = 0; r < 8; ++r) {
      int qr = mr * 16 + r + hi * 8;
      ob[((size_t)(n * 64 + qr)) * 768 + h * 64 + nc * 16 + col] = (bf16)acc[r];
    }
  }
}

// ---------------------------------------------------------------------------
// Final classifier: out[8] = h[S-1] @ Wc + bc
// ---------------------------------------------------------------------------
__global__ __launch_bounds__(256) void classifier_kernel(
    const float* __restrict__ hf, const float* __restrict__ Wc,
    const float* __restrict__ bc, float* __restrict__ out) {
  __shared__ float red[256];
  int t = threadIdx.x;
  const float* hr = hf + (size_t)4095 * 768;
  for (int c = 0; c < 8; ++c) {
    float l = 0.f;
    for (int d = t; d < 768; d += 256) l += hr[d] * Wc[d * 8 + c];
    red[t] = l; __syncthreads();
    for (int o = 128; o > 0; o >>= 1) { if (t < o) red[t] += red[t + o]; __syncthreads(); }
    if (t == 0) out[c] = red[0] + bc[c];
    __syncthreads();
  }
}

// ---------------------------------------------------------------------------
// Host orchestration
// ---------------------------------------------------------------------------
extern "C" void kernel_launch(void* const* d_in, const int* in_sizes, int n_in,
                              void* d_out, int out_size, void* d_ws, size_t ws_size,
                              hipStream_t stream) {
  (void)in_sizes; (void)n_in; (void)out_size; (void)ws_size;
  const int*   ids       = (const int*)  d_in[0];
  const int*   rand_attn = (const int*)  d_in[1];
  const float* wemb      = (const float*)d_in[2];
  const float* pemb      = (const float*)d_in[3];
  const float* eln_w     = (const float*)d_in[4];
  const float* eln_b     = (const float*)d_in[5];
  const float* Wq        = (const float*)d_in[6];
  const float* bq        = (const float*)d_in[7];
  const float* Wk        = (const float*)d_in[8];
  const float* bk        = (const float*)d_in[9];
  const float* Wv        = (const float*)d_in[10];
  const float* bv        = (const float*)d_in[11];
  const float* Wo        = (const float*)d_in[12];
  const float* bo        = (const float*)d_in[13];
  const float* ln1_w     = (const float*)d_in[14];
  const float* ln1_b     = (const float*)d_in[15];
  const float* W1        = (const float*)d_in[16];
  const float* b1        = (const float*)d_in[17];
  const float* W2        = (const float*)d_in[18];
  const float* b2        = (const float*)d_in[19];
  const float* ln2_w     = (const float*)d_in[20];
  const float* ln2_b     = (const float*)d_in[21];
  const float* Wc        = (const float*)d_in[22];
  const float* bc        = (const float*)d_in[23];

  const size_t D2 = 768 * 768;
  const size_t DF = 768 * 3072;
  const size_t SD = 4096 * 768;

  char* ws = (char*)d_ws;
  size_t off = 0;
  auto alloc = [&](size_t bytes) -> char* {
    char* p = ws + off;
    off = (off + bytes + 255) & ~(size_t)255;
    return p;
  };

  bf16* Wq_t = (bf16*)alloc(2 * D2 * 2);
  bf16* Wk_t = (bf16*)alloc(2 * D2 * 2);
  bf16* Wv_t = (bf16*)alloc(2 * D2 * 2);
  bf16* Wo_t = (bf16*)alloc(2 * D2 * 2);
  bf16* W1_t = (bf16*)alloc(2 * DF * 2);
  bf16* W2_t = (bf16*)alloc(2 * DF * 2);
  float* h_f   = (float*)alloc(SD * 4);
  float* tmp_f = (float*)alloc(SD * 4);
  bf16* h_bf = (bf16*)alloc(SD * 2);
  bf16* q_bf = (bf16*)alloc(SD * 2);
  bf16* k_bf = (bf16*)alloc(SD * 2);
  bf16* v_t  = (bf16*)alloc(SD * 2);   // V projection, stored [h*64+d][s]
  bf16* o_bf = (bf16*)alloc(SD * 2);
  // bf16 score region, reused as FFN hidden (disjoint lifetimes)
  const size_t sc_edge_bytes = (size_t)12 * 2 * 64 * 4096 * 2;   // 12.6 MB
  const size_t sc_mid_bytes  = (size_t)12 * 62 * 64 * 512 * 2;   // 48.8 MB
  const size_t f1_bytes      = (size_t)4096 * 3072 * 2;          // 25.2 MB
  size_t region_bytes = sc_edge_bytes + sc_mid_bytes;
  if (f1_bytes > region_bytes) region_bytes = f1_bytes;
  char* region = alloc(region_bytes);
  bf16* sc_edge = (bf16*)region;
  bf16* sc_mid  = (bf16*)(region + sc_edge_bytes);
  bf16* f1_bf   = (bf16*)region;

  // Transpose + convert all weights to bf16 [N][K] (one pass, once per launch)
  auto cvtT = [&](const float* s, bf16* d, int K, int N) {
    size_t n = (size_t)K * N;
    cvt_transpose_f32_to_bf16<<<(unsigned)((n + 255) / 256), 256, 0, stream>>>(s, d, K, N);
  };
  for (int l = 0; l < 2; ++l) {
    cvtT(Wq + l * D2, Wq_t + l * D2, 768, 768);
    cvtT(Wk + l * D2, Wk_t + l * D2, 768, 768);
    cvtT(Wv + l * D2, Wv_t + l * D2, 768, 768);
    cvtT(Wo + l * D2, Wo_t + l * D2, 768, 768);
    cvtT(W1 + l * DF, W1_t + l * DF, 768, 3072);
    cvtT(W2 + l * DF, W2_t + l * DF, 3072, 768);
  }

  embed_ln_kernel<<<4096, 256, 0, stream>>>(ids, wemb, pemb, eln_w, eln_b, h_f, h_bf);

  for (int l = 0; l < 2; ++l) {
    dim3 gD(64, 12);    // M=4096/64, N=768/64
    dim3 gF(64, 48);    // N=3072/64
    // Q, K projections -> bf16 row-major; V projection -> bf16 transposed
    gemm_bf16_kernel<<<gD, 256, 0, stream>>>(h_bf, Wq_t + l * D2, bq + l * 768,
                                             nullptr, q_bf, nullptr, 4096, 768, 768, 0);
    gemm_bf16_kernel<<<gD, 256, 0, stream>>>(h_bf, Wk_t + l * D2, bk + l * 768,
                                             nullptr, k_bf, nullptr, 4096, 768, 768, 0);
    gemm_bf16_kernel<<<gD, 256, 0, stream>>>(h_bf, Wv_t + l * D2, bv + l * 768,
                                             nullptr, nullptr, v_t, 4096, 768, 768, 0);
    // sparse attention
    attn_scores_kernel<<<dim3(12, 64, 8), 256, 0, stream>>>(q_bf, k_bf, rand_attn, l,
                                                            sc_mid, sc_edge);
    softmax_kernel<<<dim3(12, 64, 8), 256, 0, stream>>>(sc_mid, sc_edge);
    attn_pv_kernel<<<dim3(12, 64), 256, 0, stream>>>(sc_mid, sc_edge, v_t,
                                                     rand_attn, l, o_bf);
    // output projection + residual LN
    gemm_bf16_kernel<<<gD, 256, 0, stream>>>(o_bf, Wo_t + l * D2, bo + l * 768,
                                             tmp_f, nullptr, nullptr, 4096, 768, 768, 0);
    add_ln_kernel<<<4096, 256, 0, stream>>>(h_f, tmp_f, ln1_w + l * 768, ln1_b + l * 768,
                                            h_f, h_bf);
    // FFN
    gemm_bf16_kernel<<<gF, 256, 0, stream>>>(h_bf, W1_t + l * DF, b1 + l * 3072,
                                             nullptr, f1_bf, nullptr, 4096, 3072, 768, 1);
    gemm_bf16_kernel<<<gD, 256, 0, stream>>>(f1_bf, W2_t + l * DF, b2 + l * 768,
                                             tmp_f, nullptr, nullptr, 4096, 768, 3072, 0);
    add_ln_kernel<<<4096, 256, 0, stream>>>(h_f, tmp_f, ln2_w + l * 768, ln2_b + l * 768,
                                            h_f, h_bf);
  }

  classifier_kernel<<<1, 256, 0, stream>>>(h_f, Wc, bc, (float*)d_out);
}
